// Model_43739946942805
// MI455X (gfx1250) — matrix operand order repaired and checked by reading, not compiled, not run on hardware
//
#include <hip/hip_runtime.h>
#include <hip/hip_bf16.h>

#define BB   8
#define EE   2048
#define DD   512
#define DA   64
#define TM   16

typedef __attribute__((ext_vector_type(16))) __bf16 v16bf;
typedef __attribute__((ext_vector_type(8)))  __bf16 v8bf;
typedef __attribute__((ext_vector_type(4)))  __bf16 v4bf;
typedef __attribute__((ext_vector_type(8)))  float  v8f;

// A-fragment (16x32 bf16, row-major source, row stride = rs elements).
// lanes 0-15: M=lane, K={0..15}; lanes 16-31: M=lane-16, K={16..31}
__device__ __forceinline__ v16bf load_a_bf16(const __bf16* rowbase, int k0, int lane) {
    const __bf16* p = rowbase + k0 + ((lane & 16) ? 16 : 0);
    v8bf lo = *(const v8bf*)p;
    v8bf hi = *(const v8bf*)(p + 8);
    v16bf r;
#pragma unroll
    for (int i = 0; i < 8; i++) { r[i] = lo[i]; r[i + 8] = hi[i]; }
    return r;
}

// A-fragment built from f32 LDS row (convert to bf16 on the fly)
__device__ __forceinline__ v16bf cvt_a_from_f32(const float* rowbase, int k0, int lane) {
    const float* p = rowbase + k0 + ((lane & 16) ? 16 : 0);
    float4 a0 = *(const float4*)(p);
    float4 a1 = *(const float4*)(p + 4);
    float4 b0 = *(const float4*)(p + 8);
    float4 b1 = *(const float4*)(p + 12);
    v16bf r;
    r[0]  = (__bf16)a0.x; r[1]  = (__bf16)a0.y; r[2]  = (__bf16)a0.z; r[3]  = (__bf16)a0.w;
    r[4]  = (__bf16)a1.x; r[5]  = (__bf16)a1.y; r[6]  = (__bf16)a1.z; r[7]  = (__bf16)a1.w;
    r[8]  = (__bf16)b0.x; r[9]  = (__bf16)b0.y; r[10] = (__bf16)b0.z; r[11] = (__bf16)b0.w;
    r[12] = (__bf16)b1.x; r[13] = (__bf16)b1.y; r[14] = (__bf16)b1.z; r[15] = (__bf16)b1.w;
    return r;
}

#define WMMA_BF16(a, b, c) \
    __builtin_amdgcn_wmma_f32_16x16x32_bf16(false, (a), false, (b), (short)0, (c), false, false)

// ---------------------------------------------------------------------------
// Kernel 0: convert weights (transposed) and aux tensors to bf16
// ---------------------------------------------------------------------------
__global__ __launch_bounds__(256) void prep_cvt(
    const float* __restrict__ Wq, const float* __restrict__ Wk, const float* __restrict__ Wv,
    const float* __restrict__ qa, const float* __restrict__ ka,
    __bf16* __restrict__ WqT, __bf16* __restrict__ WkT, __bf16* __restrict__ WvT,
    __bf16* __restrict__ qab, __bf16* __restrict__ kab)
{
    const int WN = DD * DD;            // 262144
    const int AN = BB * EE * DA;       // 1048576
    int i = blockIdx.x * blockDim.x + threadIdx.x;
    if (i < 3 * WN) {
        int m = i / WN, j = i % WN;
        int n = j >> 9, k = j & 511;   // WT[n][k] = W[k][n]
        const float* W = (m == 0) ? Wq : ((m == 1) ? Wk : Wv);
        __bf16* WT     = (m == 0) ? WqT : ((m == 1) ? WkT : WvT);
        WT[j] = (__bf16)W[k * DD + n];
    } else {
        int j = i - 3 * WN;
        if (j < AN)            qab[j]      = (__bf16)qa[j];
        else if (j < 2 * AN)   kab[j - AN] = (__bf16)ka[j - AN];
    }
}

// ---------------------------------------------------------------------------
// Kernel 1: q/k/v projections (bf16 WMMA, f32 accum, bias add)
// stores q,k row-major bf16 and v transposed (d-major) bf16
// ---------------------------------------------------------------------------
__global__ __launch_bounds__(256) void qkv_proj(
    const float* __restrict__ x,
    const __bf16* __restrict__ WqT, const __bf16* __restrict__ WkT, const __bf16* __restrict__ WvT,
    const float* __restrict__ bq, const float* __restrict__ bk, const float* __restrict__ bv,
    __bf16* __restrict__ qb, __bf16* __restrict__ kb, __bf16* __restrict__ vtb)
{
    __shared__ __bf16 xs[TM * DD];     // 16 KB : x tile as bf16
    __shared__ __bf16 vls[DD * TM];    // 16 KB : v tile, d-major, for transpose

    const int b    = blockIdx.y;
    const int e0   = blockIdx.x * TM;
    const int tid  = threadIdx.x;
    const int lane = tid & 31;
    const int wave = tid >> 5;

    // stage x tile -> bf16 LDS
    {
        const int r  = tid >> 4;
        const int c0 = (tid & 15) * 32;
        const float* src = x + ((size_t)(b * EE + e0 + r)) * DD + c0;
        __bf16* dst = xs + r * DD + c0;
#pragma unroll
        for (int i = 0; i < 8; i++) {
            float4 f = *(const float4*)(src + i * 4);
            dst[i * 4 + 0] = (__bf16)f.x;
            dst[i * 4 + 1] = (__bf16)f.y;
            dst[i * 4 + 2] = (__bf16)f.z;
            dst[i * 4 + 3] = (__bf16)f.w;
        }
    }
    __syncthreads();

    const int d0    = wave * 64;            // each wave: 64 output columns
    const int n     = lane & 15;
    const int koffB = (lane & 16) ? 16 : 0;
    const __bf16* arow = xs + (lane & 15) * DD;

    v8f aq[4] = {}, ak2[4] = {}, av[4] = {};
#pragma unroll
    for (int k0 = 0; k0 < DD; k0 += 32) {
        v16bf a = load_a_bf16(arow, k0, lane);
#pragma unroll
        for (int t = 0; t < 4; t++) {
            const size_t wb = (size_t)(d0 + t * 16 + n) * DD + k0 + koffB;
            v16bf bqf = *(const v16bf*)(WqT + wb);
            v16bf bkf = *(const v16bf*)(WkT + wb);
            v16bf bvf = *(const v16bf*)(WvT + wb);
            aq[t]  = WMMA_BF16(a, bqf, aq[t]);
            ak2[t] = WMMA_BF16(a, bkf, ak2[t]);
            av[t]  = WMMA_BF16(a, bvf, av[t]);
        }
    }

    const int mbase = (lane & 16) ? 8 : 0;
#pragma unroll
    for (int t = 0; t < 4; t++) {
        const int dd = d0 + t * 16 + n;
        const float bqv = bq[dd], bkv = bk[dd], bvv = bv[dd];
#pragma unroll
        for (int r = 0; r < 8; r++) {
            const int M = r + mbase;
            const size_t o = ((size_t)(b * EE + e0 + M)) * DD + dd;
            qb[o] = (__bf16)(aq[t][r] + bqv);
            kb[o] = (__bf16)(ak2[t][r] + bkv);
            vls[dd * TM + M] = (__bf16)(av[t][r] + bvv);
        }
    }
    __syncthreads();

    // write v transposed: vT[b][d][e0..e0+15]
#pragma unroll
    for (int i = 0; i < 2; i++) {
        const int dd = tid * 2 + i;
        *(v16bf*)(vtb + ((size_t)(b * DD + dd)) * EE + e0) = *(const v16bf*)(vls + dd * TM);
    }
}

// ---------------------------------------------------------------------------
// Kernel 2: fused attention for 16 query rows per workgroup.
// Phase A: scores (main + aux) via bf16 WMMA -> blend/mask -> f32 LDS
// Phase B: softmax in LDS
// Phase C: attn @ v via bf16 WMMA (A from LDS, B from vT)
// ---------------------------------------------------------------------------
__global__ __launch_bounds__(256) void fused_attn(
    const __bf16* __restrict__ qb, const __bf16* __restrict__ kb,
    const __bf16* __restrict__ vtb,
    const __bf16* __restrict__ qab, const __bf16* __restrict__ kab,
    const int* __restrict__ amask, const float* __restrict__ mcoef,
    const float* __restrict__ thr_p, float* __restrict__ out)
{
    extern __shared__ char smem[];
    float*  sc  = (float*)smem;                                   // [16][2048] f32
    __bf16* qs  = (__bf16*)(smem + TM * EE * sizeof(float));      // [16][512]  bf16
    __bf16* qas = (__bf16*)(smem + TM * EE * 4 + TM * DD * 2);    // [16][64]   bf16

    const int b    = blockIdx.y;
    const int e0   = blockIdx.x * TM;
    const int tid  = threadIdx.x;
    const int lane = tid & 31;
    const int wave = tid >> 5;
    const float thr = *thr_p;

    // stage q rows
    {
        const int r  = tid >> 4;
        const int c0 = (tid & 15) * 32;
        const __bf16* src = qb + ((size_t)(b * EE + e0 + r)) * DD + c0;
        __bf16* dst = qs + r * DD + c0;
        *(v16bf*)dst        = *(const v16bf*)src;
        *(v16bf*)(dst + 16) = *(const v16bf*)(src + 16);
    }
    // stage query_aux rows
    {
        const int r  = tid >> 4;
        const int c0 = (tid & 15) * 4;
        *(v4bf*)(qas + r * DA + c0) =
            *(const v4bf*)(qab + ((size_t)(b * EE + e0 + r)) * DA + c0);
    }
    __syncthreads();

    const float inv_sqrt_d  = 0.044194173824159216f;  // 1/sqrt(512)
    const float inv_sqrt_da = 0.125f;                 // 1/sqrt(64)
    const size_t bEE = (size_t)b * EE * EE;
    const int n     = lane & 15;
    const int koffB = (lane & 16) ? 16 : 0;
    const int mbase = (lane & 16) ? 8 : 0;
    const __bf16* arow  = qs  + (lane & 15) * DD;
    const __bf16* aarow = qas + (lane & 15) * DA;

    // ---- Phase A: scores ----
    for (int ft = wave; ft < EE / 16; ft += 8) {
        const int f0 = ft * 16;
        v8f acc = {};
        const __bf16* krow = kb + ((size_t)(b * EE + f0 + n)) * DD + koffB;
#pragma unroll
        for (int k0 = 0; k0 < DD; k0 += 32) {
            v16bf a  = load_a_bf16(arow, k0, lane);
            v16bf bf = *(const v16bf*)(krow + k0);
            acc = WMMA_BF16(a, bf, acc);
        }
        v8f acc2 = {};
        const __bf16* karow = kab + ((size_t)(b * EE + f0 + n)) * DA + koffB;
#pragma unroll
        for (int k0 = 0; k0 < DA; k0 += 32) {
            v16bf a  = load_a_bf16(aarow, k0, lane);
            v16bf bf = *(const v16bf*)(karow + k0);
            acc2 = WMMA_BF16(a, bf, acc2);
        }
#pragma unroll
        for (int r = 0; r < 8; r++) {
            const int M = r + mbase;
            const size_t idx = bEE + (size_t)(e0 + M) * EE + (f0 + n);
            const float mc  = mcoef[idx];
            const int   adj = amask[idx];
            float sv = acc[r]  * inv_sqrt_d;
            float sa = acc2[r] * inv_sqrt_da;
            const float bl = (1.0f - mc) * sv + mc * sa;
            sv = ((sv > thr) && (sa != 0.0f)) ? bl : sv;
            sv = (adj == 0) ? -3.402823466e38f : sv;
            sc[M * EE + f0 + n] = sv;
        }
    }
    __syncthreads();

    // ---- Phase B: softmax (each wave: 2 rows) ----
    for (int row = wave; row < TM; row += 8) {
        float* s = sc + row * EE;
        float m = -3.402823466e38f;
        for (int j = lane; j < EE; j += 32) m = fmaxf(m, s[j]);
#pragma unroll
        for (int o = 16; o > 0; o >>= 1) m = fmaxf(m, __shfl_xor(m, o, 32));
        float sum = 0.0f;
        for (int j = lane; j < EE; j += 32) {
            float p = __expf(s[j] - m);
            s[j] = p;
            sum += p;
        }
#pragma unroll
        for (int o = 16; o > 0; o >>= 1) sum += __shfl_xor(sum, o, 32);
        const float inv = 1.0f / sum;
        for (int j = lane; j < EE; j += 32) s[j] *= inv;
    }
    __syncthreads();

    // ---- Phase C: out = attn @ v (each wave: 64 output columns) ----
    {
        const int d0 = wave * 64;
        const float* atrow = sc + (lane & 15) * EE;
        const __bf16* vrow = vtb + ((size_t)(b * DD + d0 + n)) * EE + koffB;
        v8f acc[4] = {};
        for (int f0 = 0; f0 < EE; f0 += 32) {
            v16bf a = cvt_a_from_f32(atrow, f0, lane);
#pragma unroll
            for (int t = 0; t < 4; t++) {
                v16bf bf = *(const v16bf*)(vrow + (size_t)t * 16 * EE + f0);
                acc[t] = WMMA_BF16(a, bf, acc[t]);
            }
        }
#pragma unroll
        for (int t = 0; t < 4; t++) {
#pragma unroll
            for (int r = 0; r < 8; r++) {
                const int M = r + mbase;
                out[((size_t)(b * EE + e0 + M)) * DD + d0 + t * 16 + n] = acc[t][r];
            }
        }
    }
}

// ---------------------------------------------------------------------------
extern "C" void kernel_launch(void* const* d_in, const int* in_sizes, int n_in,
                              void* d_out, int out_size, void* d_ws, size_t ws_size,
                              hipStream_t stream) {
    const float* x   = (const float*)d_in[0];
    const float* qa  = (const float*)d_in[1];
    const float* ka  = (const float*)d_in[2];
    const int*   adj = (const int*)  d_in[3];
    const float* mc  = (const float*)d_in[4];
    const float* Wq  = (const float*)d_in[5];
    const float* bq  = (const float*)d_in[6];
    const float* Wk  = (const float*)d_in[7];
    const float* bk  = (const float*)d_in[8];
    const float* Wv  = (const float*)d_in[9];
    const float* bv  = (const float*)d_in[10];
    const float* thr = (const float*)d_in[11];
    float* out = (float*)d_out;

    char* ws = (char*)d_ws;
    __bf16* WqT  = (__bf16*)ws;  ws += (size_t)DD * DD * 2;
    __bf16* WkT  = (__bf16*)ws;  ws += (size_t)DD * DD * 2;
    __bf16* WvT  = (__bf16*)ws;  ws += (size_t)DD * DD * 2;
    __bf16* qab  = (__bf16*)ws;  ws += (size_t)BB * EE * DA * 2;
    __bf16* kab  = (__bf16*)ws;  ws += (size_t)BB * EE * DA * 2;
    __bf16* qb   = (__bf16*)ws;  ws += (size_t)BB * EE * DD * 2;
    __bf16* kb   = (__bf16*)ws;  ws += (size_t)BB * EE * DD * 2;
    __bf16* vtb  = (__bf16*)ws;  ws += (size_t)BB * EE * DD * 2;

    {
        const int total = 3 * DD * DD + 2 * BB * EE * DA;
        const int blocks = (total + 255) / 256;
        prep_cvt<<<blocks, 256, 0, stream>>>(Wq, Wk, Wv, qa, ka, WqT, WkT, WvT, qab, kab);
    }

    qkv_proj<<<dim3(EE / TM, BB), 256, 0, stream>>>(x, WqT, WkT, WvT, bq, bk, bv, qb, kb, vtb);

    const size_t shmem = (size_t)TM * EE * 4 + (size_t)TM * DD * 2 + (size_t)TM * DA * 2; // 149504 B
    hipFuncSetAttribute(reinterpret_cast<const void*>(fused_attn),
                        hipFuncAttributeMaxDynamicSharedMemorySize, (int)shmem);
    fused_attn<<<dim3(EE / TM, BB), 256, shmem, stream>>>(
        qb, kb, vtb, qab, kab, adj, mc, thr, out);
}
